// FullGCN_52286931861996
// MI455X (gfx1250) — matrix operand order repaired and checked by reading, not compiled, not run on hardware
//
#include <hip/hip_runtime.h>
#include <hip/hip_bf16.h>

typedef __attribute__((ext_vector_type(16))) _Float16 v16h;
typedef __attribute__((ext_vector_type(8)))  float    v8f;

#define NPG   50
#define EPG   400
#define GPW   16                 // graphs per workgroup
#define NODES_W (NPG*GPW)        // 800
#define EDGES_W (EPG*GPW)        // 6400
#define BN_EPS 1e-5f

union FragH { v16h h; unsigned int u[8]; };

// ---------------------------------------------------------------------------
// kprep: fold the rank-1 GCN algebra into P^T (16x64), Q^T (16x64), constC(16);
// zero BN stat accumulators (96 floats).
// P/Q are stored TRANSPOSED: P16[col*64 + node], nodes 50..63 zero-padded,
// so WMMA B-fragment half-pairs are contiguous dwords in LDS.
// ---------------------------------------------------------------------------
__global__ void kprep(const float* W11, const float* W21,
                      const float* W12, const float* W22,
                      const float* W13, const float* W23,
                      const float* b11, const float* b21,
                      const float* b12, const float* b22,
                      const float* b13, const float* b23,
                      const float* fc1_w, const float* fc1_b,
                      _Float16* P16, _Float16* Q16, float* constC,
                      float* stats)
{
    __shared__ float U[12], C0[12], B2[12];
    int tid = threadIdx.x;
    if (tid < 12) {
        int k = tid >> 2, c = tid & 3;
        const float* W1 = (k==0) ? W11 : (k==1) ? W12 : W13;   // [1,4]
        const float* W2 = (k==0) ? W21 : (k==1) ? W22 : W23;   // [4,4]
        const float* b1 = (k==0) ? b11 : (k==1) ? b12 : b13;   // [4]
        const float* b2 = (k==0) ? b21 : (k==1) ? b22 : b23;   // [4]
        float u = 0.f, c0 = 0.f;
        for (int r = 0; r < 4; ++r) { u += W1[r]*W2[r*4+c]; c0 += b1[r]*W2[r*4+c]; }
        U[tid] = u; C0[tid] = c0; B2[tid] = b2[c];
    }
    __syncthreads();
    // transposed + padded: index = col*64 + node
    for (int idx = tid; idx < 16*64; idx += blockDim.x) {
        int col = idx >> 6, node = idx & 63;
        float p = 0.f, q = 0.f;
        if (node < NPG) {
            for (int f = 0; f < 12; ++f) {
                float w = fc1_w[(node*12 + f)*16 + col];
                p += U[f]*w; q += C0[f]*w;
            }
        }
        P16[idx] = (_Float16)p;
        Q16[idx] = (_Float16)q;
    }
    if (tid < 16) {
        float s = fc1_b[tid];
        for (int node = 0; node < NPG; ++node)
            for (int f = 0; f < 12; ++f)
                s += B2[f]*fc1_w[(node*12 + f)*16 + tid];
        constC[tid] = s;
    }
    for (int i = tid; i < 96; i += blockDim.x) stats[i] = 0.f;
}

// ---------------------------------------------------------------------------
// k1: 16 graphs / workgroup. LDS-resident scalar GCN, then FC1 via WMMA.
// ---------------------------------------------------------------------------
__global__ void __launch_bounds__(256) k1(
    const float* __restrict__ x, const int* __restrict__ ei, int E,
    const _Float16* __restrict__ P16, const _Float16* __restrict__ Q16,
    const float* __restrict__ constC,
    float* __restrict__ a1, float* stats)
{
    __shared__ unsigned int s_edges[EDGES_W];   // 25.6 KB (packed src|dst<<16, wg-local)
    __shared__ float s_x[NODES_W];
    __shared__ float s_dinv[NODES_W];           // also degree accumulator
    __shared__ float s_agg[NODES_W];
    __shared__ float s_S1[NODES_W];
    __shared__ float s_S0[NODES_W];
    __shared__ __align__(16) _Float16 s_A1[16*64];   // padded f16 S1 tile [graph][K]
    __shared__ __align__(16) _Float16 s_A0[16*64];   // padded f16 S0 tile
    __shared__ __align__(16) _Float16 s_P[16*64];    // P^T  [col][K]
    __shared__ __align__(16) _Float16 s_Q[16*64];    // Q^T  [col][K]
    __shared__ float s_red[64];
    __shared__ float s_cc[16];

    const int tid = threadIdx.x;
    const int wg  = blockIdx.x;
    const int nodeBase = wg * NODES_W;
    const int edgeBase = wg * EDGES_W;

    for (int i = tid; i < NODES_W; i += blockDim.x) {
        s_x[i] = x[nodeBase + i];
        s_dinv[i] = 0.f;
    }
    for (int i = tid; i < 16*64; i += blockDim.x) { s_P[i] = P16[i]; s_Q[i] = Q16[i]; }
    if (tid < 16) s_cc[tid] = constC[tid];
    __syncthreads();

    // single HBM pass over this WG's edge slice; count degrees
    for (int e = tid; e < EDGES_W; e += blockDim.x) {
        int s = ei[edgeBase + e]     - nodeBase;
        int d = ei[E + edgeBase + e] - nodeBase;
        s_edges[e] = (unsigned)s | ((unsigned)d << 16);
        atomicAdd(&s_dinv[d], 1.0f);
    }
    __syncthreads();
    for (int i = tid; i < NODES_W; i += blockDim.x)
        s_dinv[i] = rsqrtf(s_dinv[i] + 1.0f);     // +1 self-loop
    __syncthreads();
    // agg1 = symmetric-normalized aggregation of x (self-loop folded into init)
    for (int i = tid; i < NODES_W; i += blockDim.x) {
        float di = s_dinv[i];
        s_agg[i] = s_x[i]*di*di;
    }
    __syncthreads();
    for (int e = tid; e < EDGES_W; e += blockDim.x) {
        unsigned p = s_edges[e];
        int s = p & 0xffff, d = p >> 16;
        atomicAdd(&s_agg[d], s_x[s]*s_dinv[s]*s_dinv[d]);
    }
    __syncthreads();
    // S1 = agg of agg1; S0 = agg of ones
    for (int i = tid; i < NODES_W; i += blockDim.x) {
        float d2 = s_dinv[i]*s_dinv[i];
        s_S1[i] = s_agg[i]*d2;
        s_S0[i] = d2;
    }
    __syncthreads();
    for (int e = tid; e < EDGES_W; e += blockDim.x) {
        unsigned p = s_edges[e];
        int s = p & 0xffff, d = p >> 16;
        float w = s_dinv[s]*s_dinv[d];
        atomicAdd(&s_S1[d], s_agg[s]*w);
        atomicAdd(&s_S0[d], w);
    }
    __syncthreads();

    // pack zero-padded f16 A tiles [16 graphs][64 K] (all 256 threads)
    for (int idx = tid; idx < 16*64; idx += blockDim.x) {
        int row = idx >> 6, k = idx & 63;
        float v1 = (k < NPG) ? s_S1[row*NPG + k] : 0.f;
        float v0 = (k < NPG) ? s_S0[row*NPG + k] : 0.f;
        s_A1[idx] = (_Float16)v1;
        s_A0[idx] = (_Float16)v0;
    }
    __syncthreads();

    // FC1: out[16x16] = S1[16x50]@P + S0[16x50]@Q + constC  (wave 0, 4 WMMAs)
    if (tid < 32) {
        const int lane = tid, mrow = lane & 15, hl = lane >> 4, ncol = lane & 15;
        v8f acc = {};
        #pragma unroll
        for (int srcm = 0; srcm < 2; ++srcm) {
            const unsigned int* Ap = (const unsigned int*)((srcm == 0) ? s_A1 : s_A0);
            const unsigned int* Bp = (const unsigned int*)((srcm == 0) ? s_P  : s_Q);
            #pragma unroll
            for (int kt = 0; kt < 2; ++kt) {
                FragH a, b;
                #pragma unroll
                for (int j = 0; j < 8; ++j) {
                    int ka = kt*32 + (j>>2)*16 + (j&3)*2 + hl*8;   // even
                    int kb = kt*32 + (j>>2)*16 + (j&3)*2 + hl*8;   // even
                    a.u[j] = Ap[(mrow*64 + ka) >> 1];
                    b.u[j] = Bp[(ncol*64 + kb) >> 1];
                }
                acc = __builtin_amdgcn_wmma_f32_16x16x32_f16(
                    false, a.h, false, b.h, (short)0, acc, false, false);
            }
        }
        float cc = s_cc[ncol];
        float ssum = 0.f, ssq = 0.f;
        #pragma unroll
        for (int r = 0; r < 8; ++r) {
            float v = acc[r] + cc;
            v = v > 0.f ? v : 0.f;                                // ReLU
            int row = wg*16 + hl*8 + r;
            a1[row*16 + ncol] = v;
            ssum += v; ssq += v*v;
        }
        s_red[lane]      = ssum;
        s_red[32 + lane] = ssq;
    }
    __syncthreads();
    if (tid < 16) {
        atomicAdd(&stats[tid],      s_red[tid]      + s_red[tid + 16]);
        atomicAdd(&stats[16 + tid], s_red[32 + tid] + s_red[48 + tid]);
    }
}

// ---------------------------------------------------------------------------
// k2: BN1 affine -> FC2 (16->32) via WMMA -> ReLU -> a2 + BN2 stats.
// 64 threads = 2 full waves, wave w computes output columns [w*16, w*16+16).
// ---------------------------------------------------------------------------
__global__ void __launch_bounds__(64) k2(
    const float* __restrict__ a1, float* stats,
    const float* __restrict__ bn1_g, const float* __restrict__ bn1_b,
    const float* __restrict__ fc2_w, const float* __restrict__ fc2_b,
    float* __restrict__ a2, float invB)
{
    __shared__ float sc[16], sh[16];
    __shared__ __align__(16) _Float16 sW[32*32];  // fc2_w^T: [col][K], K padded 16->32
    __shared__ __align__(16) _Float16 sY[16*32];  // BN1-affined act tile [row][K pad 32]
    __shared__ float s_sum[32], s_sq[32];

    int tid = threadIdx.x;
    if (tid < 16) {
        float m  = stats[tid]*invB;
        float v  = stats[16 + tid]*invB - m*m;
        float rs = rsqrtf(v + BN_EPS);
        sc[tid] = rs*bn1_g[tid];
        sh[tid] = bn1_b[tid] - m*sc[tid];
    }
    if (tid < 32) { s_sum[tid] = 0.f; s_sq[tid] = 0.f; }
    for (int i = tid; i < 32*32; i += blockDim.x) {
        int col = i >> 5, k = i & 31;                 // transposed store
        sW[i] = (_Float16)((k < 16) ? fc2_w[k*32 + col] : 0.f);
    }
    __syncthreads();

    const int rowBase = blockIdx.x * 16;
    for (int i = tid; i < 16*32; i += blockDim.x) {
        int row = i >> 5, k = i & 31;
        float y = 0.f;
        if (k < 16) y = a1[(rowBase + row)*16 + k]*sc[k] + sh[k];
        sY[i] = (_Float16)y;
    }
    __syncthreads();

    const int wave = tid >> 5, lane = tid & 31;
    const int mrow = lane & 15, hl = lane >> 4;
    const int ncol = wave*16 + (lane & 15);

    const unsigned int* Yp = (const unsigned int*)sY;
    const unsigned int* Wp = (const unsigned int*)sW;
    FragH a, b;
    #pragma unroll
    for (int j = 0; j < 8; ++j) {
        int ka = (j>>2)*16 + (j&3)*2 + hl*8;          // even
        int kb = (j>>2)*16 + (j&3)*2 + hl*8;          // even
        a.u[j] = Yp[(mrow*32 + ka) >> 1];
        b.u[j] = Wp[(ncol*32 + kb) >> 1];
    }
    v8f acc = {};
    acc = __builtin_amdgcn_wmma_f32_16x16x32_f16(
        false, a.h, false, b.h, (short)0, acc, false, false);

    float bias = fc2_b[ncol];
    float ssum = 0.f, ssq = 0.f;
    #pragma unroll
    for (int r = 0; r < 8; ++r) {
        float v = acc[r] + bias;
        v = v > 0.f ? v : 0.f;
        a2[(rowBase + hl*8 + r)*32 + ncol] = v;
        ssum += v; ssq += v*v;
    }
    atomicAdd(&s_sum[ncol], ssum);
    atomicAdd(&s_sq[ncol],  ssq);
    __syncthreads();
    if (tid < 32) {
        atomicAdd(&stats[32 + tid], s_sum[tid]);
        atomicAdd(&stats[64 + tid], s_sq[tid]);
    }
}

// ---------------------------------------------------------------------------
// k3: BN2 affine + FC3 (32->2) -> logits
// ---------------------------------------------------------------------------
__global__ void __launch_bounds__(256) k3(
    const float* __restrict__ a2, const float* __restrict__ stats,
    const float* __restrict__ bn2_g, const float* __restrict__ bn2_b,
    const float* __restrict__ fc3_w, const float* __restrict__ fc3_b,
    float* __restrict__ out, float invB, int B)
{
    __shared__ float sc[32], sh[32], w0[32], w1[32];
    int tid = threadIdx.x;
    if (tid < 32) {
        float m  = stats[32 + tid]*invB;
        float v  = stats[64 + tid]*invB - m*m;
        float rs = rsqrtf(v + BN_EPS);
        sc[tid] = rs*bn2_g[tid];
        sh[tid] = bn2_b[tid] - m*sc[tid];
        w0[tid] = fc3_w[tid*2 + 0];
        w1[tid] = fc3_w[tid*2 + 1];
    }
    __syncthreads();
    int row = blockIdx.x*blockDim.x + tid;
    if (row < B) {
        float o0 = fc3_b[0], o1 = fc3_b[1];
        #pragma unroll
        for (int c = 0; c < 32; ++c) {
            float y = a2[row*32 + c]*sc[c] + sh[c];
            o0 += y*w0[c];
            o1 += y*w1[c];
        }
        out[row*2 + 0] = o0;
        out[row*2 + 1] = o1;
    }
}

// ---------------------------------------------------------------------------
extern "C" void kernel_launch(void* const* d_in, const int* in_sizes, int n_in,
                              void* d_out, int out_size, void* d_ws, size_t ws_size,
                              hipStream_t stream) {
    const float* x     = (const float*)d_in[0];
    const int*   ei    = (const int*)d_in[1];
    const float* W11   = (const float*)d_in[2];
    const float* b11   = (const float*)d_in[3];
    const float* W21   = (const float*)d_in[4];
    const float* b21   = (const float*)d_in[5];
    const float* W12   = (const float*)d_in[6];
    const float* b12   = (const float*)d_in[7];
    const float* W22   = (const float*)d_in[8];
    const float* b22   = (const float*)d_in[9];
    const float* W13   = (const float*)d_in[10];
    const float* b13   = (const float*)d_in[11];
    const float* W23   = (const float*)d_in[12];
    const float* b23   = (const float*)d_in[13];
    const float* fc1_w = (const float*)d_in[14];
    const float* fc1_b = (const float*)d_in[15];
    const float* bn1_g = (const float*)d_in[16];
    const float* bn1_b = (const float*)d_in[17];
    const float* fc2_w = (const float*)d_in[18];
    const float* fc2_b = (const float*)d_in[19];
    const float* bn2_g = (const float*)d_in[20];
    const float* bn2_b = (const float*)d_in[21];
    const float* fc3_w = (const float*)d_in[22];
    const float* fc3_b = (const float*)d_in[23];

    const int N = in_sizes[0];         // B * 50
    const int B = N / NPG;             // 32768
    const int E = in_sizes[1] / 2;     // 13107200

    // workspace layout (floats)
    float* ws     = (float*)d_ws;
    float* a1     = ws;                          // B*16
    float* a2     = a1 + (size_t)B*16;           // B*32
    float* stats  = a2 + (size_t)B*32;           // 96: sum1,sq1,sum2[32],sq2[32]
    float* constC = stats + 96;                  // 16
    _Float16* P16 = (_Float16*)(constC + 16);    // 16*64 halves (transposed, padded)
    _Float16* Q16 = P16 + 16*64;                 // 16*64 halves

    const float invB = 1.0f / (float)B;

    kprep<<<1, 128, 0, stream>>>(W11, W21, W12, W22, W13, W23,
                                 b11, b21, b12, b22, b13, b23,
                                 fc1_w, fc1_b, P16, Q16, constC, stats);
    k1<<<B / GPW, 256, 0, stream>>>(x, ei, E, P16, Q16, constC, a1, stats);
    k2<<<B / 16, 64, 0, stream>>>(a1, stats, bn1_g, bn1_b, fc2_w, fc2_b, a2, invB);
    k3<<<(B + 255) / 256, 256, 0, stream>>>(a2, stats, bn2_g, bn2_b, fc3_w, fc3_b,
                                            (float*)d_out, invB, B);
}
